// MessageFunction_44160853737513
// MI455X (gfx1250) — matrix operand order repaired and checked by reading, not compiled, not run on hardware
//
#include <hip/hip_runtime.h>

#define V_NODES 50000
#define E_EDGES 800000
#define THREADS 256
#define WAVES 8

// ---- LDS layout (bytes) ----
#define OFF_PB1 0          // packed B, layer1: 4nt * 6kb * 32lanes * 32B = 24576
#define OFF_PB2 24576      // layer2: 4*3*32*32 = 12288
#define OFF_PB3 36864      // layer3: 12288
#define OFF_SW  49152      // small weights, 2433 f32 (padded to 2440) = 9760 B
#define OFF_AB  58912      // per-wave A rows: 8 waves * 16 rows * 384 B = 49152
#define OFF_SB  108064     // per-wave s-buffer: 8 * 16*64*4 = 32768
#define LDS_TOTAL 140832

typedef __attribute__((ext_vector_type(16))) __bf16 v16bf;
typedef __attribute__((ext_vector_type(8)))  float  v8f;

struct B32x8 { uint4 lo; uint4 hi; };

__device__ __forceinline__ unsigned short f2bf(float f) {
  union { float f; unsigned u; } v; v.f = f;
  unsigned r = v.u + 0x7FFFu + ((v.u >> 16) & 1u);   // round-to-nearest-even
  return (unsigned short)(r >> 16);
}
__device__ __forceinline__ unsigned f2bf_pk(float a, float b) {
  return (unsigned)f2bf(a) | ((unsigned)f2bf(b) << 16);
}
__device__ __forceinline__ v16bf ld_frag(const char* p0, const char* p1) {
  B32x8 t;
  t.lo = *(const uint4*)p0;
  t.hi = *(const uint4*)p1;
  return __builtin_bit_cast(v16bf, t);
}

// Pre-swizzle a [Kreal x 64] row-major f32 weight matrix into the WMMA
// B-matrix (32x16 bf16 per fragment) register-image layout, in LDS.
// Fragment (nt,kb): lane n<16 pair v = {W[32kb+2v][16nt+n], W[32kb+2v+1][16nt+n]},
// lane 16+n: same with +16 on K. Zero-padded past Kreal.
__device__ __forceinline__ void stage_pb(char* dst, const float* __restrict__ W,
                                         int Kreal, int nkb, int tid) {
  const int entries = 4 * nkb * 32;
  for (int idx = tid; idx < entries; idx += THREADS) {
    const int lane2 = idx & 31;
    const int kb = (idx >> 5) % nkb;
    const int nt = (idx >> 5) / nkb;
    const int col = (nt << 4) | (lane2 & 15);
    const int kbase = kb * 32 + ((lane2 & 16) ? 16 : 0);
    unsigned o[8];
#pragma unroll
    for (int v = 0; v < 8; ++v) {
      const int k0 = kbase + 2 * v;
      const float a = (k0 < Kreal)     ? W[k0 * 64 + col]       : 0.f;
      const float b = (k0 + 1 < Kreal) ? W[(k0 + 1) * 64 + col] : 0.f;
      o[v] = f2bf_pk(a, b);
    }
    uint4* p = (uint4*)(dst + (size_t)idx * 32);
    p[0] = make_uint4(o[0], o[1], o[2], o[3]);
    p[1] = make_uint4(o[4], o[5], o[6], o[7]);
  }
}

__device__ __forceinline__ void cpy(float* dst, const float* __restrict__ src, int n, int tid) {
  for (int i = tid; i < n; i += THREADS) dst[i] = src[i];
}

// 16-edge x 64-out GEMM tile: A rows (bf16, stride 384B) x packed B -> sbuf f32 (+bias).
template<int NKB>
__device__ __forceinline__ void wave_gemm(const char* aW, const char* pb,
                                          const float* bias, float* sbuf, int lane) {
  // A operand, contiguous k order used here (B packed to match): lane<16 row=lane holds K {0..15}; lane>=16 row=lane-16 holds K {16..31}
  const char* arow = aW + (size_t)(lane & 15) * 384 + ((lane & 16) ? 32 : 0);
  const int col0 = lane & 15;
  const int rbase = (lane & 16) ? 8 : 0;   // C layout: lanes 0-15 -> M=r, lanes 16-31 -> M=r+8
#pragma unroll
  for (int nt = 0; nt < 4; ++nt) {
    v8f c = {0.f, 0.f, 0.f, 0.f, 0.f, 0.f, 0.f, 0.f};
#pragma unroll
    for (int kb = 0; kb < NKB; ++kb) {
      v16bf a = ld_frag(arow + kb * 64, arow + kb * 64 + 16);
      const char* bp = pb + (size_t)((nt * NKB + kb) * 32 + lane) * 32;
      v16bf b = ld_frag(bp, bp + 16);
      c = __builtin_amdgcn_wmma_f32_16x16x32_bf16(false, a, false, b, (short)0, c,
                                                  false, false);
    }
    const int col = (nt << 4) + col0;
    const float bb = bias[col];
#pragma unroll
    for (int r = 0; r < 8; ++r)
      sbuf[(rbase + r) * 64 + col] = c[r] + bb;
  }
}

// vector channel for dv_in = h = dv_out = 8: Vh = Vin@Wh, vn = ||Vh||_R, vmu = Vh@Wmu
__device__ __forceinline__ void vchan8(const float vin[24], const float* Wh,
                                       const float* Wmu, float vn[8], float vmu[24]) {
  float vh[24];
#pragma unroll
  for (int h = 0; h < 8; ++h) {
    float a0 = 0.f, a1 = 0.f, a2 = 0.f;
#pragma unroll
    for (int d = 0; d < 8; ++d) {
      const float w = Wh[d * 8 + h];
      a0 += vin[d] * w; a1 += vin[8 + d] * w; a2 += vin[16 + d] * w;
    }
    vh[h] = a0; vh[8 + h] = a1; vh[16 + h] = a2;
    vn[h] = sqrtf(a0 * a0 + a1 * a1 + a2 * a2 + 1e-8f);
  }
#pragma unroll
  for (int i = 0; i < 24; ++i) vmu[i] = 0.f;
#pragma unroll
  for (int h = 0; h < 8; ++h)
#pragma unroll
    for (int o = 0; o < 8; ++o) {
      const float w = Wmu[h * 8 + o];
      vmu[o] += vh[h] * w; vmu[8 + o] += vh[8 + h] * w; vmu[16 + o] += vh[16 + h] * w;
    }
}

__device__ __forceinline__ void gate8(const float* srow, const float* Wg,
                                      const float* bg, float g[8]) {
  float a[8];
#pragma unroll
  for (int o = 0; o < 8; ++o) a[o] = bg[o];
  for (int c = 0; c < 64; ++c) {
    const float sv = srow[c];
#pragma unroll
    for (int o = 0; o < 8; ++o) a[o] += sv * Wg[c * 8 + o];
  }
#pragma unroll
  for (int o = 0; o < 8; ++o) g[o] = 1.f / (1.f + __expf(-a[o]));
}

__global__ void gvp_message_kernel(
    const float* __restrict__ gs, const float* __restrict__ gV,
    const float* __restrict__ gse, const float* __restrict__ gVe,
    const int* __restrict__ gei,
    const float* __restrict__ Wh1, const float* __restrict__ Ws1, const float* __restrict__ bs1,
    const float* __restrict__ Wmu1, const float* __restrict__ Wg1, const float* __restrict__ bg1,
    const float* __restrict__ Wh2, const float* __restrict__ Ws2, const float* __restrict__ bs2,
    const float* __restrict__ Wmu2, const float* __restrict__ Wg2, const float* __restrict__ bg2,
    const float* __restrict__ Wh3, const float* __restrict__ Ws3, const float* __restrict__ bs3,
    const float* __restrict__ Wmu3, const float* __restrict__ Wg3, const float* __restrict__ bg3,
    float* __restrict__ out_s, float* __restrict__ out_V) {
  extern __shared__ char smem[];
  const int tid = threadIdx.x;
  const int lane = tid & 31;
  const int wave = tid >> 5;

  // ---- block init: stage packed WMMA-B weights + small weights into LDS ----
  stage_pb(smem + OFF_PB1, Ws1, 161, 6, tid);
  stage_pb(smem + OFF_PB2, Ws2, 72, 3, tid);
  stage_pb(smem + OFF_PB3, Ws3, 72, 3, tid);
  float* sw = (float*)(smem + OFF_SW);
  cpy(sw + 0,    Wh1, 289, tid);   // 17x17
  cpy(sw + 289,  Wmu1, 136, tid);  // 17x8
  cpy(sw + 425,  Wg1, 512, tid);   // 64x8
  cpy(sw + 937,  bs1, 64, tid);
  cpy(sw + 1001, bg1, 8, tid);
  cpy(sw + 1009, Wh2, 64, tid);
  cpy(sw + 1073, Wmu2, 64, tid);
  cpy(sw + 1137, Wg2, 512, tid);
  cpy(sw + 1649, bs2, 64, tid);
  cpy(sw + 1713, bg2, 8, tid);
  cpy(sw + 1721, Wh3, 64, tid);
  cpy(sw + 1785, Wmu3, 64, tid);
  cpy(sw + 1849, Wg3, 512, tid);
  cpy(sw + 2361, bs3, 64, tid);
  cpy(sw + 2425, bg3, 8, tid);
  __syncthreads();

  const int tile = blockIdx.x * WAVES + wave;   // 16 edges per wave
  const int ge0 = tile * 16;
  const int e = lane & 15;
  const int ge = ge0 + e;
  const int src = gei[ge];
  const int dst = gei[E_EDGES + ge];

  char* aW = smem + OFF_AB + (size_t)wave * 6144;          // 16 rows x 384 B (192 bf16 cols)
  float* sbuf = (float*)(smem + OFF_SB + (size_t)wave * 4096); // 16 x 64 f32
  unsigned* arow = (unsigned*)(aW + (size_t)e * 384);      // bf16 pairs

  // ---- layer-1 A row: cols [s_src | s_dst | s_e | vn1 | 0-pad] as bf16 ----
  {
    const int node = (lane < 16) ? src : dst;
    const float* sp = gs + (size_t)node * 64;
    const int cb = (lane < 16) ? 0 : 32;       // uint index = col/2
#pragma unroll 8
    for (int j = 0; j < 32; ++j) arow[cb + j] = f2bf_pk(sp[2 * j], sp[2 * j + 1]);
  }

  float vmu1[24];
  if (lane < 16) {
    const float* sep = gse + (size_t)ge * 16;
#pragma unroll
    for (int j = 0; j < 8; ++j) arow[64 + j] = f2bf_pk(sep[2 * j], sep[2 * j + 1]);

    // vector channel layer 1: V_cat [3,17] -> Vh1 [3,17], vn1[17], Vmu1[3,8]
    float vc[3][17];
    const float* vs = gV + (size_t)src * 24;
    const float* vd = gV + (size_t)dst * 24;
#pragma unroll
    for (int r = 0; r < 3; ++r) {
#pragma unroll
      for (int d = 0; d < 8; ++d) { vc[r][d] = vs[r * 8 + d]; vc[r][8 + d] = vd[r * 8 + d]; }
      vc[r][16] = gVe[(size_t)ge * 3 + r];
    }
    float vh[3][17];
    const float* wh = sw;
    for (int h = 0; h < 17; ++h) {
      float a0 = 0.f, a1 = 0.f, a2 = 0.f;
      for (int d = 0; d < 17; ++d) {
        const float w = wh[d * 17 + h];
        a0 += vc[0][d] * w; a1 += vc[1][d] * w; a2 += vc[2][d] * w;
      }
      vh[0][h] = a0; vh[1][h] = a1; vh[2][h] = a2;
    }
    float vn[17];
    for (int h = 0; h < 17; ++h)
      vn[h] = sqrtf(vh[0][h] * vh[0][h] + vh[1][h] * vh[1][h] + vh[2][h] * vh[2][h] + 1e-8f);
#pragma unroll
    for (int i = 0; i < 24; ++i) vmu1[i] = 0.f;
    const float* wmu = sw + 289;
    for (int h = 0; h < 17; ++h)
#pragma unroll
      for (int o = 0; o < 8; ++o) {
        const float w = wmu[h * 8 + o];
        vmu1[o] += vh[0][h] * w; vmu1[8 + o] += vh[1][h] * w; vmu1[16 + o] += vh[2][h] * w;
      }
#pragma unroll
    for (int j = 0; j < 8; ++j) arow[72 + j] = f2bf_pk(vn[2 * j], vn[2 * j + 1]);
    arow[80] = f2bf_pk(vn[16], 0.f);
#pragma unroll
    for (int j = 81; j < 96; ++j) arow[j] = 0u;   // cols 162..191 pad
  }
  __syncthreads();

  wave_gemm<6>(aW, smem + OFF_PB1, sw + 937, sbuf, lane);   // s1_lin (pre-relu) -> sbuf
  __syncthreads();

  // ---- layer-1 gate/gating + build layer-2 A row ----
  float vmu2[24];
  if (lane < 16) {
    const float* srow = sbuf + e * 64;
    float g[8];
    gate8(srow, sw + 425, sw + 1001, g);
    float v1[24];
#pragma unroll
    for (int i = 0; i < 24; ++i) v1[i] = vmu1[i] * g[i & 7];
    float vn2[8];
    vchan8(v1, sw + 1009, sw + 1073, vn2, vmu2);
#pragma unroll 8
    for (int j = 0; j < 32; ++j) {
      float a = srow[2 * j], b = srow[2 * j + 1];
      a = a > 0.f ? a : 0.f; b = b > 0.f ? b : 0.f;     // relu for next-layer input
      arow[j] = f2bf_pk(a, b);
    }
#pragma unroll
    for (int j = 0; j < 4; ++j) arow[32 + j] = f2bf_pk(vn2[2 * j], vn2[2 * j + 1]);
#pragma unroll
    for (int j = 36; j < 48; ++j) arow[j] = 0u;       // cols 72..95 pad
  }
  __syncthreads();

  wave_gemm<3>(aW, smem + OFF_PB2, sw + 1649, sbuf, lane);
  __syncthreads();

  // ---- layer-2 gate/gating + build layer-3 A row ----
  float vmu3[24];
  if (lane < 16) {
    const float* srow = sbuf + e * 64;
    float g[8];
    gate8(srow, sw + 1137, sw + 1713, g);
    float v2[24];
#pragma unroll
    for (int i = 0; i < 24; ++i) v2[i] = vmu2[i] * g[i & 7];
    float vn3[8];
    vchan8(v2, sw + 1721, sw + 1785, vn3, vmu3);
#pragma unroll 8
    for (int j = 0; j < 32; ++j) {
      float a = srow[2 * j], b = srow[2 * j + 1];
      a = a > 0.f ? a : 0.f; b = b > 0.f ? b : 0.f;
      arow[j] = f2bf_pk(a, b);
    }
#pragma unroll
    for (int j = 0; j < 4; ++j) arow[32 + j] = f2bf_pk(vn3[2 * j], vn3[2 * j + 1]);
#pragma unroll
    for (int j = 36; j < 48; ++j) arow[j] = 0u;
  }
  __syncthreads();

  wave_gemm<3>(aW, smem + OFF_PB3, sw + 2361, sbuf, lane);  // s3 (no relu) -> sbuf
  __syncthreads();

  // ---- layer-3 gate + V3 output ----
  if (lane < 16) {
    const float* srow = sbuf + e * 64;
    float g[8];
    gate8(srow, sw + 1849, sw + 2425, g);
    float* vout = out_V + (size_t)ge * 24;
#pragma unroll
    for (int i = 0; i < 24; ++i) vout[i] = vmu3[i] * g[i & 7];
  }
  // ---- s3 output: contiguous 16x64 f32 block, coalesced float4 copy ----
  {
    const float4* s4 = (const float4*)sbuf;
    float4* d4 = (float4*)(out_s + (size_t)ge0 * 64);
    for (int i = lane; i < 256; i += 32) d4[i] = s4[i];
  }
}

extern "C" void kernel_launch(void* const* d_in, const int* in_sizes, int n_in,
                              void* d_out, int out_size, void* d_ws, size_t ws_size,
                              hipStream_t stream) {
  const float* gs  = (const float*)d_in[0];
  const float* gV  = (const float*)d_in[1];
  const float* gse = (const float*)d_in[2];
  const float* gVe = (const float*)d_in[3];
  const int*   gei = (const int*)d_in[4];
  const float* W[18];
  for (int i = 0; i < 18; ++i) W[i] = (const float*)d_in[5 + i];

  float* out_s = (float*)d_out;                       // [E, 64]
  float* out_V = out_s + (size_t)E_EDGES * 64;        // [E, 3, 8]

  dim3 grid(E_EDGES / (16 * WAVES));                  // 6250 blocks, 8 wave-tiles each
  gvp_message_kernel<<<grid, THREADS, LDS_TOTAL, stream>>>(
      gs, gV, gse, gVe, gei,
      W[0],  W[1],  W[2],  W[3],  W[4],  W[5],
      W[6],  W[7],  W[8],  W[9],  W[10], W[11],
      W[12], W[13], W[14], W[15], W[16], W[17],
      out_s, out_V);
}
